// OneLayerGCN_17824114279163
// MI455X (gfx1250) — matrix operand order repaired and checked by reading, not compiled, not run on hardware
//
#include <hip/hip_runtime.h>

// Problem constants (match reference)
#define B_SUB   4096
#define S_SUB   64
#define N_NODES (B_SUB * S_SUB)
#define E_EDGES 2097152
#define INF_    300
#define OUTF_   64
#define K_PAD   320
#define KK_STEPS 10          // K_PAD / 32
#define EPB     512          // edges per subgraph = E / B
#define PACK_U32 (KK_STEPS * 2 * 64 * 8)   // 10240 u32 per packed operand
#define TILE_ELTS (S_SUB * INF_)           // 19200 f32 per subgraph feature tile

// LDS layout (bytes). Staging aliases sB/sFeat/sAgg (dead after A-pack phase).
#define SA_OFF     0            // 40960 : A fragments (bf16 packed)
#define R2_OFF     40960        // staging: 76800 raw f32 tile  (TDM target)
#define SB_OFF     40960        // 40960 : B fragments (aliases staging)
#define SFEAT_OFF  81920        // 16640 : GEMM result, stride 65
#define SAGG_OFF   98560        // 16640 : SpMM accumulator
#define SMALL_OFF  117760       // norms/bias/pool/misc
#define SMEM_BYTES 118800

typedef __attribute__((ext_vector_type(16))) __bf16 v16bf;
typedef __attribute__((ext_vector_type(8)))  float  v8f;
typedef __attribute__((ext_vector_type(4)))  unsigned int u32x4;
typedef __attribute__((ext_vector_type(8)))  int    i32x8;
typedef __attribute__((ext_vector_type(4)))  int    i32x4;

__device__ __forceinline__ unsigned bf16_rn(float f) {
    unsigned x = __float_as_uint(f);
    return (x + 0x7FFFu + ((x >> 16) & 1u)) >> 16;   // round-to-nearest-even
}
__device__ __forceinline__ unsigned pack2(float lo, float hi) {
    return bf16_rn(lo) | (bf16_rn(hi) << 16);
}

struct U32x8 { uint4 lo, hi; };
__device__ __forceinline__ v16bf ld_frag(const unsigned* p) {
    U32x8 u;
    u.lo = *(const uint4*)p;        // ds_load_b128
    u.hi = *(const uint4*)(p + 4);  // ds_load_b128
    return __builtin_bit_cast(v16bf, u);
}

// wave32 xor-swizzle add (group-of-32 mode: offset = xor<<10 | and_mask 0x1F)
#define SWZ_ADD(v, mask) ((v) + __uint_as_float((unsigned)__builtin_amdgcn_ds_swizzle( \
        (int)__float_as_uint(v), (((mask) << 10) | 0x1F))))

// ---------------- helper kernels ----------------

__global__ void zero_u32_kernel(unsigned* __restrict__ p, int n) {
    int i = blockIdx.x * blockDim.x + threadIdx.x;
    if (i < n) p[i] = 0u;
}

__global__ void degree_kernel(const int* __restrict__ src, const int* __restrict__ dst,
                              unsigned* __restrict__ degS, unsigned* __restrict__ degD) {
    int e = blockIdx.x * blockDim.x + threadIdx.x;
    if (e < E_EDGES) {
        atomicAdd(&degS[src[e]], 1u);
        atomicAdd(&degD[dst[e]], 1u);
    }
}

// Pre-pack weight [300,64] -> bf16 in the exact WMMA B-fragment (32x16 bf16) lane
// layout, K zero-padded to 320. Consumer reads 8 contiguous u32 per lane.
__global__ void pack_weight_kernel(const float* __restrict__ w, unsigned* __restrict__ bp) {
    int idx = blockIdx.x * blockDim.x + threadIdx.x;
    if (idx >= PACK_U32) return;
    int v  = idx & 7;
    int n  = (idx >> 3) & 63;
    int hh = (idx >> 9) & 1;
    int kk = idx >> 10;
    int ke = kk * 32 + hh * 16 + 2 * v;
    float f0 = (ke     < INF_) ? w[ke * OUTF_ + n]       : 0.f;
    float f1 = (ke + 1 < INF_) ? w[(ke + 1) * OUTF_ + n] : 0.f;
    bp[idx] = pack2(f0, f1);
}

// ---------------- fused per-subgraph kernel ----------------

__global__ __launch_bounds__(256)
void gcn_fused_kernel(const float* __restrict__ in_feat,
                      const float* __restrict__ edge_weight,
                      const float* __restrict__ bias,
                      const float* __restrict__ prelu_a,
                      const int*   __restrict__ src,
                      const int*   __restrict__ dst,
                      const unsigned* __restrict__ degS,
                      const unsigned* __restrict__ degD,
                      const unsigned* __restrict__ Bpack,
                      float* __restrict__ outH,
                      float* __restrict__ outPool,
                      float* __restrict__ outAnchor)
{
    extern __shared__ char smem_raw[];
    unsigned* sA   = (unsigned*)(smem_raw + SA_OFF);
    float* sStage  = (float*)(smem_raw + R2_OFF);     // raw f32 tile (TDM dest)
    unsigned* sB   = (unsigned*)(smem_raw + SB_OFF);  // aliases staging
    float* sFeat   = (float*)(smem_raw + SFEAT_OFF);  // aliases staging
    float* sAgg    = (float*)(smem_raw + SAGG_OFF);   // aliases staging
    float* sNsrc   = (float*)(smem_raw + SMALL_OFF);
    float* sNdst   = sNsrc + 64;
    float* sBias   = sNdst + 64;
    float* sPool   = sBias + 64;
    float* sMisc   = sPool + 64;

    const int tid = threadIdx.x;
    const int b   = blockIdx.x;
    const float pa = prelu_a[0];

    // -------- Phase 0: wave 0 issues TDM DMA of the contiguous 76.8 KB feature
    // tile (64 rows x 300 f32) into LDS staging; other waves set up norms. ----
    if (tid < 32) {
        const float* gsrc = in_feat + (size_t)b * S_SUB * INF_;
        unsigned long long ga = (unsigned long long)(uintptr_t)gsrc;
        unsigned lds_addr = (unsigned)(uintptr_t)(void*)sStage;

        u32x4 g0;
        g0[0] = 1u;                                   // count=1, user descriptor
        g0[1] = lds_addr;                             // LDS dest (bytes)
        g0[2] = (unsigned)(ga & 0xFFFFFFFFull);       // global_addr[31:0]
        g0[3] = (unsigned)((ga >> 32) & 0x01FFFFFFull) | (2u << 30); // [56:32]|type=2

        i32x8 g1;
        g1[0] = (int)(2u << 16);                      // mask=0, data_size=4B
        g1[1] = (int)((TILE_ELTS & 0xFFFFu) << 16);   // tensor_dim0[15:0]
        g1[2] = (int)(((unsigned)TILE_ELTS >> 16) | (1u << 16)); // dim0 hi | dim1=1
        g1[3] = (int)((TILE_ELTS & 0xFFFFu) << 16);   // tile_dim0 = 19200 (1D)
        g1[4] = 0;                                    // tile_dim1=0 (unused), tile_dim2=0
        g1[5] = (int)TILE_ELTS;                       // tensor_dim0_stride
        g1[6] = 0;
        g1[7] = 0;

        i32x4 z4 = {0, 0, 0, 0};
#if defined(__clang_major__) && (__clang_major__ >= 23)
        i32x8 z8 = {0, 0, 0, 0, 0, 0, 0, 0};
        __builtin_amdgcn_tensor_load_to_lds(g0, g1, z4, z4, z8, 0);
#else
        __builtin_amdgcn_tensor_load_to_lds(g0, g1, z4, z4, 0);
#endif
        __builtin_amdgcn_s_wait_tensorcnt(0);         // issuing wave waits; barrier publishes
    }
    if (tid >= 64 && tid < 128) {
        int t = tid - 64;
        int g = b * S_SUB + t;
        sNsrc[t] = rsqrtf(fmaxf((float)degS[g], 1.f));
        sNdst[t] = rsqrtf(fmaxf((float)degD[g], 1.f));
        sBias[t] = bias[t];
        sPool[t] = 0.f;
        if (t == 0) sMisc[0] = sqrtf(fmaxf((float)degS[b * S_SUB], 1.f));
    }
    __syncthreads();

    // -------- Phase 1: build A-pack from LDS staging: scale by nsrc, cvt bf16,
    // A-fragment (16x32) in this code's contiguous k order (same as B pack): lanes 0-15 K{0..15}, 16-31 K{16..31}.
    for (int idx = tid; idx < PACK_U32; idx += 256) {
        int v  = idx & 7;
        int m  = (idx >> 3) & 63;
        int hh = (idx >> 9) & 1;
        int kk = idx >> 10;
        int ke = kk * 32 + hh * 16 + 2 * v;
        float s = sNsrc[m];
        const float* row = sStage + m * INF_;
        float f0 = (ke     < INF_) ? row[ke] * s     : 0.f;
        float f1 = (ke + 1 < INF_) ? row[ke + 1] * s : 0.f;
        sA[idx] = pack2(f0, f1);
    }
    __syncthreads();   // staging is dead past this point

    // -------- Phase 2: stage pre-packed weights (overwrites staging) + zero agg
    {
        const uint4* gb = (const uint4*)Bpack;
        uint4* lb = (uint4*)sB;
        for (int i = tid; i < PACK_U32 / 4; i += 256) lb[i] = gb[i];
    }
    for (int i = tid; i < 64 * 65; i += 256) sAgg[i] = 0.f;
    __syncthreads();

    // -------- GEMM: feat[64,64] = Ahat @ W via v_wmma_f32_16x16x32_bf16 --------
    const int lane = tid & 31;
    const int wv   = tid >> 5;      // wave 0..7
    const int hl   = lane >> 4;     // lane half
    const int l16  = lane & 15;
    const int tm   = wv >> 1;       // output tile row 0..3
    const int tn0  = (wv & 1) * 2;  // two adjacent tile cols per wave

    v8f c0 = {0.f, 0.f, 0.f, 0.f, 0.f, 0.f, 0.f, 0.f};
    v8f c1 = c0;
    #pragma unroll
    for (int kk = 0; kk < KK_STEPS; ++kk) {
        const unsigned base = (unsigned)((kk * 2 + hl) * 64) * 8u;
        v16bf af  = ld_frag(&sA[base + (unsigned)(tm * 16 + l16) * 8u]);
        v16bf bf0 = ld_frag(&sB[base + (unsigned)(tn0 * 16 + l16) * 8u]);
        v16bf bf1 = ld_frag(&sB[base + (unsigned)((tn0 + 1) * 16 + l16) * 8u]);
        c0 = __builtin_amdgcn_wmma_f32_16x16x32_bf16(false, af, false, bf0,
                                                     (short)0, c0, false, false);
        c1 = __builtin_amdgcn_wmma_f32_16x16x32_bf16(false, af, false, bf1,
                                                     (short)0, c1, false, false);
    }
    // C layout: VGPR r -> (M = r + 8*half, N = lane%16)
    #pragma unroll
    for (int r = 0; r < 8; ++r) {
        int m = tm * 16 + hl * 8 + r;
        sFeat[m * 65 + tn0 * 16 + l16]       = c0[r];
        sFeat[m * 65 + (tn0 + 1) * 16 + l16] = c1[r];
    }
    __syncthreads();

    // -------- SpMM: agg[d] += feat[s] * ew, 512 local edges, LDS float atomics ----
    const int q  = tid & 3;        // feature quarter
    const int f0 = q * 16;
    #pragma unroll 1
    for (int it = 0; it < EPB / 64; ++it) {
        int e = b * EPB + (tid >> 2) + it * 64;
        int s = src[e] - b * S_SUB;
        int d = dst[e] - b * S_SUB;
        float w = edge_weight[e];
        const float* fs = &sFeat[s * 65 + f0];
        float* ad = &sAgg[d * 65 + f0];
        #pragma unroll
        for (int f = 0; f < 16; ++f) atomicAdd(&ad[f], fs[f] * w);
    }
    __syncthreads();

    // -------- h = prelu(agg * ndst + bias); pool accumulate; row L2 norm --------
    {
        const int m = tid >> 2;
        const float nd = sNdst[m];
        float hv[16];
        float ss = 0.f;
        #pragma unroll
        for (int f = 0; f < 16; ++f) {
            float x = sAgg[m * 65 + f0 + f] * nd + sBias[f0 + f];
            x = (x >= 0.f) ? x : pa * x;
            hv[f] = x;
            ss += x * x;
            atomicAdd(&sPool[f0 + f], x);
        }
        ss = SWZ_ADD(ss, 1);     // reduce across the 4 lanes owning row m
        ss = SWZ_ADD(ss, 2);
        float inv = 1.f / fmaxf(sqrtf(ss), 1e-12f);
        float* o = outH + (size_t)(b * S_SUB + m) * OUTF_ + f0;
        #pragma unroll
        for (int f = 0; f < 16; ++f) o[f] = hv[f] * inv;
    }
    __syncthreads();

    if (tid < 32) {
        // pool = mean over 64 rows of post-prelu h, then L2 norm
        float p0 = sPool[tid]      * (1.f / 64.f);
        float p1 = sPool[tid + 32] * (1.f / 64.f);
        float ss = p0 * p0 + p1 * p1;
        ss = SWZ_ADD(ss, 1); ss = SWZ_ADD(ss, 2); ss = SWZ_ADD(ss, 4);
        ss = SWZ_ADD(ss, 8); ss = SWZ_ADD(ss, 16);
        float inv = 1.f / fmaxf(sqrtf(ss), 1e-12f);
        outPool[b * OUTF_ + tid]      = p0 * inv;
        outPool[b * OUTF_ + tid + 32] = p1 * inv;
    } else if (tid < 64) {
        // anchor: undo the src-norm on GEMM row 0 -> raw in_feat[0] @ W
        int l = tid - 32;
        float s0 = sMisc[0];
        float a0 = sFeat[l]      * s0 + sBias[l];
        float a1 = sFeat[l + 32] * s0 + sBias[l + 32];
        a0 = (a0 >= 0.f) ? a0 : pa * a0;
        a1 = (a1 >= 0.f) ? a1 : pa * a1;
        float ss = a0 * a0 + a1 * a1;
        ss = SWZ_ADD(ss, 1); ss = SWZ_ADD(ss, 2); ss = SWZ_ADD(ss, 4);
        ss = SWZ_ADD(ss, 8); ss = SWZ_ADD(ss, 16);
        float inv = 1.f / fmaxf(sqrtf(ss), 1e-12f);
        outAnchor[b * OUTF_ + l]      = a0 * inv;
        outAnchor[b * OUTF_ + l + 32] = a1 * inv;
    }
}

// ---------------- launch ----------------

extern "C" void kernel_launch(void* const* d_in, const int* in_sizes, int n_in,
                              void* d_out, int out_size, void* d_ws, size_t ws_size,
                              hipStream_t stream) {
    (void)in_sizes; (void)n_in; (void)out_size; (void)ws_size;

    const float* in_feat     = (const float*)d_in[0];
    const float* edge_weight = (const float*)d_in[1];
    const float* weight      = (const float*)d_in[2];
    const float* bias        = (const float*)d_in[3];
    const float* prelu_a     = (const float*)d_in[4];
    const int*   src         = (const int*)d_in[5];
    const int*   dst         = (const int*)d_in[6];

    float* outH      = (float*)d_out;                       // N*64
    float* outPool   = outH + (size_t)N_NODES * OUTF_;      // B*64
    float* outAnchor = outPool + (size_t)B_SUB * OUTF_;     // B*64

    unsigned* degS  = (unsigned*)d_ws;        // N
    unsigned* degD  = degS + N_NODES;         // N
    unsigned* Bpack = degD + N_NODES;         // 10240

    zero_u32_kernel<<<(2 * N_NODES + 255) / 256, 256, 0, stream>>>(degS, 2 * N_NODES);
    degree_kernel<<<E_EDGES / 256, 256, 0, stream>>>(src, dst, degS, degD);
    pack_weight_kernel<<<(PACK_U32 + 255) / 256, 256, 0, stream>>>(weight, Bpack);

    (void)hipFuncSetAttribute(reinterpret_cast<const void*>(gcn_fused_kernel),
                              hipFuncAttributeMaxDynamicSharedMemorySize, SMEM_BYTES);

    gcn_fused_kernel<<<B_SUB, 256, SMEM_BYTES, stream>>>(
        in_feat, edge_weight, bias, prelu_a, src, dst,
        degS, degD, Bpack, outH, outPool, outAnchor);
}
